// MultiHeadAttention_19842748908205
// MI455X (gfx1250) — compile-verified, matrix operand order checked by reading, not run on hardware
//
#include <hip/hip_runtime.h>
#include <hip/hip_bf16.h>

typedef __attribute__((ext_vector_type(16))) _Float16 v16h;
typedef __attribute__((ext_vector_type(8)))  float    v8f;

static constexpr int Bq = 4;
static constexpr int Sq = 1024;
static constexpr int Hq = 16;
static constexpr int Dq = 64;
static constexpr int MD = 1024;

// ---------------------------------------------------------------------------
// Transpose + convert: Wt[n*K + k] = (f16) W[k*N + n].  32x32 tiles via LDS
// so both the f32 reads and the f16 writes are coalesced.
// ---------------------------------------------------------------------------
__global__ __launch_bounds__(256) void transpose_to_f16(
    const float* __restrict__ W, _Float16* __restrict__ Wt, int K, int N)
{
    __shared__ float tile[32][33];
    const int n0 = blockIdx.x << 5;
    const int k0 = blockIdx.y << 5;
    const int tx = threadIdx.x & 31;
    const int ty = threadIdx.x >> 5;              // 0..7
#pragma unroll
    for (int i = ty; i < 32; i += 8)
        tile[i][tx] = W[(size_t)(k0 + i) * N + n0 + tx];
    __syncthreads();
#pragma unroll
    for (int i = ty; i < 32; i += 8)
        Wt[(size_t)(n0 + i) * K + k0 + tx] = (_Float16)tile[tx][i];
}

// ---------------------------------------------------------------------------
// GEMM with bias: Y = (X @ Wt^T + bias) * scale
//   X : (M,K) TIn row-major.  Wt: (N,K) f16 (pre-transposed) -> B-fragment
//   reads are contiguous.  The 16-row A strip (16 x K) is shared by all 8
//   waves of the block, so it is staged into LDS once with CDNA5 async
//   global->LDS copies (ASYNCcnt), then fragments are read from LDS.
//   TRANS_OUT stores Y as (B, N, Sdim) per-batch-transposed (used for V).
// ---------------------------------------------------------------------------
template <typename TIn, typename TOut, bool TRANS_OUT>
__global__ __launch_bounds__(256) void wmma_gemm_bias(
    const TIn* __restrict__ X, const _Float16* __restrict__ Wt,
    const float* __restrict__ bias, TOut* __restrict__ Y,
    int M, int N, int K, float scale, int Sdim)
{
    __shared__ TIn As[16 * 1024];                  // K <= 1024 by construction

    const int lane = threadIdx.x & 31;
    const int w    = threadIdx.x >> 5;             // wave 0..7
    const int row0 = blockIdx.y << 4;              // block shares 16 rows
    const int col0 = (blockIdx.x << 9) + (w << 6); // 512 cols/block, 64/wave
    const int m    = lane & 15;
    const int half = lane >> 4;

    // --- async-stage the contiguous 16xK strip of X into LDS --------------
    {
        const unsigned stripBytes = 16u * (unsigned)K * (unsigned)sizeof(TIn);
        const char* gsrc = (const char*)(X + (size_t)row0 * K);
        const unsigned lbase = (unsigned)(uintptr_t)(&As[0]);
        for (unsigned off = threadIdx.x * 16u; off < stripBytes; off += 256u * 16u) {
            unsigned           ldst  = lbase + off;
            unsigned long long gaddr = (unsigned long long)(gsrc + off);
            asm volatile("global_load_async_to_lds_b128 %0, %1, off"
                         :: "v"(ldst), "v"(gaddr) : "memory");
        }
        asm volatile("s_wait_asynccnt 0x0" ::: "memory");
        __syncthreads();
    }

    v8f acc[4] = {};
    const TIn* arow = &As[m * K];

    for (int k0 = 0; k0 < K; k0 += 32) {
        v16h a;
#pragma unroll
        for (int j = 0; j < 8; ++j) {
            const int kk = k0 + ((j >> 2) << 4) + (half << 3) + ((j & 3) << 1);
            a[2 * j]     = (_Float16)arow[kk];     // ds_load, 16B runs
            a[2 * j + 1] = (_Float16)arow[kk + 1];
        }
        if (k0 + 32 < K)                           // uniform branch
            __builtin_prefetch(&Wt[(size_t)(col0 + m) * K + k0 + 32], 0, 0);
#pragma unroll
        for (int t = 0; t < 4; ++t) {
            const int n = col0 + (t << 4) + m;
            const _Float16* wrow = Wt + (size_t)n * K;
            v16h bf;
#pragma unroll
            for (int j = 0; j < 8; ++j) {
                const int kk = k0 + ((j >> 2) << 4) + (half << 3) + ((j & 3) << 1);
                bf[2 * j]     = wrow[kk];          // contiguous -> b128
                bf[2 * j + 1] = wrow[kk + 1];
            }
            acc[t] = __builtin_amdgcn_wmma_f32_16x16x32_f16(
                false, a, false, bf, (short)0, acc[t], false, false);
        }
    }

#pragma unroll
    for (int t = 0; t < 4; ++t) {
        const int n = col0 + (t << 4) + m;
        const float bv = bias[n];
#pragma unroll
        for (int r = 0; r < 8; ++r) {
            const int mm = row0 + r + (half << 3);
            const float val = (acc[t][r] + bv) * scale;
            if (TRANS_OUT) {
                const int b = mm / Sdim, s = mm % Sdim;
                Y[((size_t)b * N + n) * Sdim + s] = (TOut)val;
            } else {
                Y[(size_t)mm * N + n] = (TOut)val;
            }
        }
    }
}

// ---------------------------------------------------------------------------
// Logits + mask + softmax. One block per (b, h, 16-row strip).
// 8 waves x 16x128 logit chunks via WMMA (K = depth = 64), masked logits into
// 64 KB LDS, then wave-level shfl softmax, f32 attn rows to d_out.
// Qh already carries the 1/64 logit scale.
// ---------------------------------------------------------------------------
__global__ __launch_bounds__(256) void attn_logits_softmax(
    const _Float16* __restrict__ Qh, const _Float16* __restrict__ Kh,
    const float* __restrict__ mask, float* __restrict__ attn)
{
    __shared__ float lds[16 * Sq];                 // 64 KB

    const int lane = threadIdx.x & 31;
    const int w    = threadIdx.x >> 5;
    const int rowBlocks = Sq >> 4;
    const int bh   = blockIdx.x / rowBlocks;
    const int rb   = blockIdx.x % rowBlocks;
    const int b    = bh / Hq;
    const int h    = bh % Hq;
    const int row0 = rb << 4;
    const int m    = lane & 15;
    const int half = lane >> 4;

    const _Float16* qrow = Qh + ((size_t)(b * Sq + row0 + m) * MD + h * Dq);

    v8f acc[8] = {};
    const int colBase = w << 7;

    for (int k0 = 0; k0 < Dq; k0 += 32) {
        v16h a;
#pragma unroll
        for (int j = 0; j < 8; ++j) {
            const int kk = k0 + ((j >> 2) << 4) + (half << 3) + ((j & 3) << 1);
            a[2 * j]     = qrow[kk];
            a[2 * j + 1] = qrow[kk + 1];
        }
#pragma unroll
        for (int t = 0; t < 8; ++t) {
            const int n = colBase + (t << 4) + m;
            const _Float16* krow = Kh + ((size_t)(b * Sq + n) * MD + h * Dq);
            v16h bf;
#pragma unroll
            for (int j = 0; j < 8; ++j) {
                const int kk = k0 + ((j >> 2) << 4) + (half << 3) + ((j & 3) << 1);
                bf[2 * j]     = krow[kk];
                bf[2 * j + 1] = krow[kk + 1];
            }
            acc[t] = __builtin_amdgcn_wmma_f32_16x16x32_f16(
                false, a, false, bf, (short)0, acc[t], false, false);
        }
    }

    const float* mrow = mask + (size_t)b * Sq;
#pragma unroll
    for (int t = 0; t < 8; ++t) {
        const int n = colBase + (t << 4) + m;
        const float madd = mrow[n] * -1e9f;
#pragma unroll
        for (int r = 0; r < 8; ++r)
            lds[(r + (half << 3)) * Sq + n] = acc[t][r] + madd;
    }
    __syncthreads();

    for (int rr = 0; rr < 2; ++rr) {
        const int row = (w << 1) + rr;
        float* lrow = lds + row * Sq;

        float mx = -1e30f;
        for (int n = lane; n < Sq; n += 32) mx = fmaxf(mx, lrow[n]);
#pragma unroll
        for (int off = 16; off > 0; off >>= 1) mx = fmaxf(mx, __shfl_xor(mx, off, 32));

        float sum = 0.f;
        for (int n = lane; n < Sq; n += 32) {
            const float p = __expf(lrow[n] - mx);
            lrow[n] = p;
            sum += p;
        }
#pragma unroll
        for (int off = 16; off > 0; off >>= 1) sum += __shfl_xor(sum, off, 32);
        const float inv = 1.0f / sum;

        float* orow = attn + ((size_t)bh * Sq + row0 + row) * Sq;
        for (int n = lane; n < Sq; n += 32) orow[n] = lrow[n] * inv;
    }
}

// ---------------------------------------------------------------------------
// ctx = attn @ V.  One wave per 16x64 tile (full head depth), K = S = 1024.
// A fragments converted f32->f16 from attn; B fragments from the per-head
// transposed Vt (B, MD, S) so the k-pairs are contiguous -> b128 loads.
// ---------------------------------------------------------------------------
__global__ __launch_bounds__(256) void attn_av(
    const float* __restrict__ attn, const _Float16* __restrict__ Vt,
    _Float16* __restrict__ ctx)
{
    const int lane = threadIdx.x & 31;
    const int wid  = (blockIdx.x * blockDim.x + threadIdx.x) >> 5;
    const int rowBlocks = Sq >> 4;
    const int bh   = wid / rowBlocks;
    const int rb   = wid % rowBlocks;
    const int b    = bh / Hq;
    const int h    = bh % Hq;
    const int row0 = rb << 4;
    const int m    = lane & 15;
    const int half = lane >> 4;

    const float* arow = attn + ((size_t)bh * Sq + row0 + m) * Sq;

    v8f acc[4] = {};

    for (int k0 = 0; k0 < Sq; k0 += 32) {
        v16h a;
#pragma unroll
        for (int j = 0; j < 8; ++j) {
            const int kk = k0 + ((j >> 2) << 4) + (half << 3) + ((j & 3) << 1);
            a[2 * j]     = (_Float16)arow[kk];
            a[2 * j + 1] = (_Float16)arow[kk + 1];
        }
#pragma unroll
        for (int t = 0; t < 4; ++t) {
            const int n = (t << 4) + m;
            const _Float16* vrow = Vt + ((size_t)b * MD + h * Dq + n) * Sq;
            v16h bf;
#pragma unroll
            for (int j = 0; j < 8; ++j) {
                const int kk = k0 + ((j >> 2) << 4) + (half << 3) + ((j & 3) << 1);
                bf[2 * j]     = vrow[kk];          // contiguous -> b128
                bf[2 * j + 1] = vrow[kk + 1];
            }
            acc[t] = __builtin_amdgcn_wmma_f32_16x16x32_f16(
                false, a, false, bf, (short)0, acc[t], false, false);
        }
    }

#pragma unroll
    for (int t = 0; t < 4; ++t) {
        const int n = (t << 4) + m;
#pragma unroll
        for (int r = 0; r < 8; ++r) {
            const int mm = row0 + r + (half << 3);
            ctx[(size_t)(b * Sq + mm) * MD + h * Dq + n] = (_Float16)acc[t][r];
        }
    }
}

// ---------------------------------------------------------------------------
extern "C" void kernel_launch(void* const* d_in, const int* in_sizes, int n_in,
                              void* d_out, int out_size, void* d_ws, size_t ws_size,
                              hipStream_t stream)
{
    const float* q    = (const float*)d_in[0];
    const float* k    = (const float*)d_in[1];
    const float* v    = (const float*)d_in[2];
    const float* mask = (const float*)d_in[3];
    const float* Wq   = (const float*)d_in[4];
    const float* bq   = (const float*)d_in[5];
    const float* Wk   = (const float*)d_in[6];
    const float* bk   = (const float*)d_in[7];
    const float* Wv   = (const float*)d_in[8];
    const float* bv   = (const float*)d_in[9];
    const float* Wo   = (const float*)d_in[10];
    const float* bo   = (const float*)d_in[11];

    float* out  = (float*)d_out;                         // (B,S,MD)
    float* attn = out + (size_t)Bq * Sq * MD;            // (B,H,S,S)

    const size_t elems = (size_t)Bq * Sq * MD;           // 4M
    const size_t welems = (size_t)MD * MD;               // 1M
    _Float16* Qh  = (_Float16*)d_ws;
    _Float16* Kh  = Qh + elems;
    _Float16* Vt  = Kh + elems;                          // (B, MD, S)
    _Float16* ctx = Vt + elems;
    _Float16* Wqt = ctx + elems;                         // (N,K) f16 each
    _Float16* Wkt = Wqt + welems;
    _Float16* Wvt = Wkt + welems;
    _Float16* Wot = Wvt + welems;                        // total ws: 40 MB

    const int M = Bq * Sq;                               // 4096
    const dim3 blk(256);

    // Weight transpose + f16 convert (one-time, ~16 MB traffic)
    const dim3 tgrid(MD / 32, MD / 32);
    transpose_to_f16<<<tgrid, blk, 0, stream>>>(Wq, Wqt, MD, MD);
    transpose_to_f16<<<tgrid, blk, 0, stream>>>(Wk, Wkt, MD, MD);
    transpose_to_f16<<<tgrid, blk, 0, stream>>>(Wv, Wvt, MD, MD);
    transpose_to_f16<<<tgrid, blk, 0, stream>>>(Wo, Wot, MD, MD);

    // Projections: grid (N/512, M/16) = (2, 256)
    const dim3 ggrid(MD / 512, M / 16);
    wmma_gemm_bias<float, _Float16, false><<<ggrid, blk, 0, stream>>>(
        q, Wqt, bq, Qh, M, MD, MD, 1.0f / (float)Dq, Sq);
    wmma_gemm_bias<float, _Float16, false><<<ggrid, blk, 0, stream>>>(
        k, Wkt, bk, Kh, M, MD, MD, 1.0f, Sq);
    wmma_gemm_bias<float, _Float16, true><<<ggrid, blk, 0, stream>>>(
        v, Wvt, bv, Vt, M, MD, MD, 1.0f, Sq);

    // Logits + mask + softmax -> attn (f32 output tensor)
    attn_logits_softmax<<<Bq * Hq * (Sq / 16), blk, 0, stream>>>(Qh, Kh, mask, attn);

    // ctx = attn @ V
    attn_av<<<Bq * Hq * (Sq / 16) / 8, blk, 0, stream>>>(attn, Vt, ctx);

    // out = ctx @ Wo + bo
    wmma_gemm_bias<_Float16, float, false><<<ggrid, blk, 0, stream>>>(
        ctx, Wot, bo, out, M, MD, MD, 1.0f, Sq);
}
